// SC_18657337934487
// MI455X (gfx1250) — matrix operand order repaired and checked by reading, not compiled, not run on hardware
//
#include <hip/hip_runtime.h>

// ---------------------------------------------------------------------------
// LISTA sparse coding for MI455X (gfx1250, wave32, WMMA + TDM).
//   y  = x @ Dict                      (50000x1024x512, bf16 WMMA, f32 acc)
//   z0 = soft(y, l);  z = soft(z @ S + y/c, l)  x5   (50000x512x512 each)
//   l  = MLP(x)/c  (layer1 via WMMA, LN+softplus+small layers in LDS)
//   c  = lambda_max(Dict^T Dict) via power iteration (G symmetric => S = I-G/c)
// B matrices (Dict, S, w1) are pre-packed (kt-major) into the wave32 bf16
// B-fragment layout; per-K-step 16KB slabs are DMA'd into double-buffered LDS
// with TENSOR_LOAD_TO_LDS (tracked by TENSORcnt) and consumed via ds_load.
// A tiles are converted f32->bf16 on the fly per the 16-bit A-fragment layout.
// ---------------------------------------------------------------------------

typedef __attribute__((ext_vector_type(16))) __bf16 v16bf;
typedef __attribute__((ext_vector_type(8)))  float  v8f;
typedef __attribute__((ext_vector_type(4)))  float  v4f;

#define NROWS   50000
#define LDIM    1024
#define NATOMS  512
#define TITERS  5
#define MODE_FIRST 0
#define MODE_ITER  1

#if __has_builtin(__builtin_amdgcn_tensor_load_to_lds) && \
    __has_builtin(__builtin_amdgcn_s_wait_tensorcnt)
#define HAVE_TDM 1
#else
#define HAVE_TDM 0
#endif

#if HAVE_TDM
typedef __attribute__((ext_vector_type(4))) unsigned int u32x4;
typedef __attribute__((ext_vector_type(8))) int          i32x8;
typedef __attribute__((ext_vector_type(4))) int          i32x4;

// 1-D contiguous TDM load: `elems` bf16 elements from gaddr -> LDS lds_addr.
// D# group0: count=1, lds_addr, 57-bit global addr, type=2 ("image").
// D# group1: data_size=1 (2B), tensor_dim0=tile_dim0=elems, tile_dim1=1.
// Groups 2/3 (and the trailing group of the 6-arg builtin) are zero: 1-D tile.
__device__ __forceinline__ void tdm_load_slab(unsigned lds_addr,
                                              unsigned long long gaddr,
                                              unsigned elems) {
  u32x4 g0;
  g0.x = 1u;                                                // count=1
  g0.y = lds_addr;                                          // LDS byte offset
  g0.z = (unsigned)(gaddr & 0xffffffffu);
  g0.w = (unsigned)((gaddr >> 32) & 0x1ffffffu) | (2u << 30); // addr hi, type=2
  i32x8 g1;
  g1[0] = (int)(1u << 16);                      // data_size=1 -> 2 bytes
  g1[1] = (int)((elems & 0xffffu) << 16);       // tensor_dim0[15:0]  @bits48
  g1[2] = (int)((elems >> 16) | (1u << 16));    // tensor_dim0[31:16], t_dim1=1
  g1[3] = (int)(elems << 16);                   // tile_dim0          @bits112
  g1[4] = 1;                                    // tile_dim1=1, tile_dim2=0
  g1[5] = (int)elems;                           // tensor_dim0_stride (benign)
  g1[6] = 0;
  g1[7] = 0;
  i32x4 zz4 = {0, 0, 0, 0};
  i32x8 zz8 = {0, 0, 0, 0, 0, 0, 0, 0};
  // 6-arg form (clang-23 / therock-10.0 headers): (g0, g1, g2, g3, g4, cpol)
  __builtin_amdgcn_tensor_load_to_lds(g0, g1, zz4, zz4, zz8, 0);
}
#endif

__device__ __forceinline__ v8f wmma_bf16(v16bf a, v16bf b, v8f c) {
  return __builtin_amdgcn_wmma_f32_16x16x32_bf16(false, a, false, b,
                                                 (short)0, c, false, false);
}

// A fragment (16x32 bf16) from row-major f32. Lane holds M = lane&15,
// K = {c0..c0+7, c0+16..c0+23}, c0 = (lane<16 ? 0 : 8).
__device__ __forceinline__ v16bf load_a_frag(const float* __restrict__ Abase,
                                             int kk, int lane) {
  const float* p = Abase + kk + ((lane & 16) >> 1);
  v4f a0 = *(const v4f*)(p);
  v4f a1 = *(const v4f*)(p + 4);
  v4f b0 = *(const v4f*)(p + 16);
  v4f b1 = *(const v4f*)(p + 20);
  v16bf r;
  r[0]  = (__bf16)a0[0]; r[1]  = (__bf16)a0[1]; r[2]  = (__bf16)a0[2]; r[3]  = (__bf16)a0[3];
  r[4]  = (__bf16)a1[0]; r[5]  = (__bf16)a1[1]; r[6]  = (__bf16)a1[2]; r[7]  = (__bf16)a1[3];
  r[8]  = (__bf16)b0[0]; r[9]  = (__bf16)b0[1]; r[10] = (__bf16)b0[2]; r[11] = (__bf16)b0[3];
  r[12] = (__bf16)b1[0]; r[13] = (__bf16)b1[1]; r[14] = (__bf16)b1[2]; r[15] = (__bf16)b1[3];
  return r;
}

// B fragment from swizzled buffer (kt-major tiles of 512 bf16, lane-major).
__device__ __forceinline__ v16bf load_b_frag(const __bf16* __restrict__ Bsw,
                                             int tile, int lane) {
  return *(const v16bf*)(Bsw + ((size_t)tile << 9) + (lane << 4));
}

__device__ __forceinline__ float softplus_f(float x) {
  return x > 20.0f ? x : log1pf(expf(x));
}

// ---------------------------------------------------------------------------
// G = Dict^T @ Dict   (512x512). One block per output row.
// ---------------------------------------------------------------------------
__global__ __launch_bounds__(256) void gram_kernel(const float* __restrict__ D,
                                                   float* __restrict__ G) {
  __shared__ float colg[LDIM];
  int g = blockIdx.x;
  for (int k = threadIdx.x; k < LDIM; k += 256) colg[k] = D[k * NATOMS + g];
  __syncthreads();
  for (int n = threadIdx.x; n < NATOMS; n += 256) {
    float s = 0.0f;
    for (int k = 0; k < LDIM; ++k) s = fmaf(colg[k], D[k * NATOMS + n], s);
    G[g * NATOMS + n] = s;
  }
}

// ---------------------------------------------------------------------------
// Power iteration: c = lambda_max(G) = ||Dict||_2^2.  cbuf = {c, 1/c}.
// ---------------------------------------------------------------------------
__global__ __launch_bounds__(512) void power_kernel(const float* __restrict__ G,
                                                    float* __restrict__ cbuf) {
  __shared__ float v[NATOMS];
  __shared__ float red[NATOMS];
  int t = threadIdx.x;
  v[t] = 1.0f;
  __syncthreads();
  float nrm = 1.0f;
  for (int it = 0; it < 64; ++it) {
    float s = 0.0f;
    const float* Gr = G + t * NATOMS;
    for (int n = 0; n < NATOMS; ++n) s = fmaf(Gr[n], v[n], s);
    red[t] = s * s;
    __syncthreads();
    for (int st = 256; st > 0; st >>= 1) {
      if (t < st) red[t] += red[t + st];
      __syncthreads();
    }
    nrm = sqrtf(red[0]);
    __syncthreads();
    v[t] = s / nrm;
    __syncthreads();
  }
  if (t == 0) { cbuf[0] = nrm; cbuf[1] = 1.0f / nrm; }
}

// ---------------------------------------------------------------------------
// Pack row-major f32 [K x N] into kt-major bf16 B-fragment tiles:
// tile = kt*(N/16) + nt; within tile, lane-major (16 elems/lane),
// element i of lane l -> K = kt*32 + ((l&16)>>1) + (i&7) + ((i&8)<<1), N = nt*16 + (l&15).
// mode 1: value = (k==n) - src[k,n]/c   (builds S = I - G/c directly)
// ---------------------------------------------------------------------------
__global__ __launch_bounds__(256) void pack_b_sw(const float* __restrict__ src,
                                                 __bf16* __restrict__ dst,
                                                 int K, int N, int mode,
                                                 const float* __restrict__ cbuf) {
  int tid = blockIdx.x * 256 + threadIdx.x;
  if (tid >= K * N) return;
  int ntiles = N >> 4;
  int tile   = tid >> 9;
  int within = tid & 511;
  int l = within >> 4;
  int i = within & 15;
  int kt = tile / ntiles;
  int nt = tile - kt * ntiles;
  int k = (kt << 5) + ((l & 16) >> 1) + (i & 7) + ((i & 8) << 1);
  int n = (nt << 4) + (l & 15);
  float v = src[k * N + n];
  if (mode == 1) v = ((k == n) ? 1.0f : 0.0f) - v * cbuf[1];
  dst[tid] = (__bf16)v;
}

// ---------------------------------------------------------------------------
// Fused MLP: l[row] = lam(x_row)/c. 64 rows/block; layer1 WMMA into LDS.
// ---------------------------------------------------------------------------
__global__ __launch_bounds__(256) void mlp_kernel(
    const float* __restrict__ x, const __bf16* __restrict__ w1sw,
    const float* __restrict__ b1, const float* __restrict__ g1,
    const float* __restrict__ beta1,
    const float* __restrict__ w2, const float* __restrict__ b2,
    const float* __restrict__ g2, const float* __restrict__ beta2,
    const float* __restrict__ w3, const float* __restrict__ b3,
    const float* __restrict__ cbuf, float* __restrict__ lout, int M) {
  __shared__ float h1[64][128];
  __shared__ float h2[64][64];
  int lane = threadIdx.x & 31;
  int wave = threadIdx.x >> 5;
  int mw = wave >> 1, nw = wave & 1;
  int m0 = blockIdx.x * 64;

  { // layer 1: [64x1024] @ [1024x128] via WMMA, wave tile 16M x 64N
    int arow = m0 + mw * 16 + (lane & 15);
    if (arow >= M) arow = M - 1;
    const float* Ab = x + (size_t)arow * LDIM;
    int nbase = nw * 64;
    v8f acc[4] = {};
    for (int kt = 0; kt < 32; ++kt) {
      int kk = kt << 5;
      __builtin_prefetch(Ab + kk + 64, 0, 0);
      v16bf a = load_a_frag(Ab, kk, lane);
#pragma unroll
      for (int nt = 0; nt < 4; ++nt) {
        v16bf b = load_b_frag(w1sw, kt * 8 + (nbase >> 4) + nt, lane);
        acc[nt] = wmma_bf16(a, b, acc[nt]);
      }
    }
#pragma unroll
    for (int nt = 0; nt < 4; ++nt) {
      int col = nbase + (nt << 4) + (lane & 15);
#pragma unroll
      for (int r = 0; r < 8; ++r) {
        int rl = mw * 16 + ((lane & 16) ? (8 + r) : r);
        h1[rl][col] = acc[nt][r] + b1[col];
      }
    }
  }
  __syncthreads();

  int t = threadIdx.x;
  if (t < 64) { // LN + softplus over 128
    float s = 0.0f, s2 = 0.0f;
    for (int j = 0; j < 128; ++j) { float h = h1[t][j]; s += h; s2 += h * h; }
    float mu  = s * (1.0f / 128.0f);
    float var = s2 * (1.0f / 128.0f) - mu * mu;
    float inv = rsqrtf(var + 1e-5f);
    for (int j = 0; j < 128; ++j) {
      float h = (h1[t][j] - mu) * inv * g1[j] + beta1[j];
      h1[t][j] = softplus_f(h);
    }
  }
  __syncthreads();

  { // layer 2: [64x128] @ [128x64]; 4 threads x 16 cols per row
    int row = threadIdx.x >> 2;
    int cg  = (threadIdx.x & 3) << 4;
    for (int c = 0; c < 16; ++c) {
      int col = cg + c;
      float s = b2[col];
      for (int j = 0; j < 128; ++j) s = fmaf(h1[row][j], w2[j * 64 + col], s);
      h2[row][col] = s;
    }
  }
  __syncthreads();

  if (t < 64) { // LN + softplus over 64, layer-3 dot, scale by 1/c
    float s = 0.0f, s2 = 0.0f;
    for (int j = 0; j < 64; ++j) { float h = h2[t][j]; s += h; s2 += h * h; }
    float mu  = s * (1.0f / 64.0f);
    float var = s2 * (1.0f / 64.0f) - mu * mu;
    float inv = rsqrtf(var + 1e-5f);
    float lam = b3[0];
    for (int j = 0; j < 64; ++j) {
      float h = (h2[t][j] - mu) * inv * g2[j] + beta2[j];
      lam = fmaf(softplus_f(h), w3[j], lam);
    }
    int row = m0 + t;
    if (row < M) lout[row] = lam * cbuf[1];
  }
}

// ---------------------------------------------------------------------------
// Main GEMM + LISTA epilogue. Block: 8 waves, tile 64M x 256N; wave 16M x 128N.
// B slabs (16KB per K-step) are TDM-DMA'd into double-buffered LDS by wave 0,
// synchronized with s_wait_tensorcnt + workgroup barriers.
// MODE_FIRST: acc = x@Dict = y; store yc = y/c, Zout = soft(y, l)
// MODE_ITER : acc = z@S;   v = acc + yc; Zout = soft(v, l)
// ---------------------------------------------------------------------------
__global__ __launch_bounds__(256) void lista_gemm(
    const float* __restrict__ A, int lda, int K,
    const __bf16* __restrict__ Bsw,
    const float* __restrict__ lthr, const float* __restrict__ cbuf,
    const float* __restrict__ yc_in, float* __restrict__ yc_out,
    float* __restrict__ Zout, int M, int mode) {
  int lane = threadIdx.x & 31;
  int wave = threadIdx.x >> 5;
  int mw = wave >> 1, nw = wave & 1;
  int m0 = blockIdx.y * 64 + mw * 16;
  int n0 = blockIdx.x * 256 + nw * 128;
  int arow = m0 + (lane & 15);
  if (arow >= M) arow = M - 1;
  const float* Abase = A + (size_t)arow * lda;
  int ktiles = K >> 5;

  v8f acc[8] = {};

#if HAVE_TDM
  // Double-buffered LDS staging of the block's B slab (16 tiles = 16KB / step).
  __shared__ __align__(32) __bf16 bsm[2][16 * 512];
  const unsigned lds0 = (unsigned)(size_t)&bsm[0][0];   // LDS byte offset
  const unsigned long long bbase = (unsigned long long)(size_t)Bsw;
  const int nt0 = blockIdx.x * 16;                      // first of 16 nt tiles
  if (wave == 0)
    tdm_load_slab(lds0, bbase + (unsigned long long)(nt0) * 1024ull, 16 * 512);
  for (int kt = 0; kt < ktiles; ++kt) {
    if (wave == 0) {
      if (kt + 1 < ktiles) {
        tdm_load_slab(lds0 + (unsigned)((kt + 1) & 1) * 16384u,
                      bbase + (unsigned long long)((kt + 1) * 32 + nt0) * 1024ull,
                      16 * 512);
        __builtin_amdgcn_s_wait_tensorcnt(1);  // slab kt complete (in-order)
      } else {
        __builtin_amdgcn_s_wait_tensorcnt(0);
      }
    }
    __syncthreads();
    __builtin_prefetch(Abase + (kt << 5) + 64, 0, 0);
    v16bf a = load_a_frag(Abase, kt << 5, lane);
    const __bf16* bs = bsm[kt & 1] + (size_t)(nw * 8) * 512 + (lane << 4);
#pragma unroll
    for (int nt = 0; nt < 8; ++nt) {
      v16bf b = *(const v16bf*)(bs + (size_t)nt * 512);
      acc[nt] = wmma_bf16(a, b, acc[nt]);
    }
    __syncthreads();  // all waves done with buf[kt&1] before it is re-filled
  }
#else
  int ntbase = n0 >> 4;
  for (int kt = 0; kt < ktiles; ++kt) {
    __builtin_prefetch(Abase + (kt << 5) + 64, 0, 0);
    v16bf a = load_a_frag(Abase, kt << 5, lane);
#pragma unroll
    for (int nt = 0; nt < 8; ++nt) {
      v16bf b = load_b_frag(Bsw, kt * 32 + ntbase + nt, lane);
      acc[nt] = wmma_bf16(a, b, acc[nt]);
    }
  }
#endif

  float inv_c = cbuf[1];
  float lv[8];
#pragma unroll
  for (int r = 0; r < 8; ++r) {
    int row = m0 + ((lane & 16) ? (8 + r) : r);
    lv[r] = lthr[row < M ? row : (M - 1)];
  }
#pragma unroll
  for (int nt = 0; nt < 8; ++nt) {
    int n = n0 + (nt << 4) + (lane & 15);
#pragma unroll
    for (int r = 0; r < 8; ++r) {
      int row = m0 + ((lane & 16) ? (8 + r) : r);
      if (row >= M) continue;
      size_t idx = (size_t)row * NATOMS + n;
      float v = acc[nt][r];
      if (mode == MODE_FIRST) {
        yc_out[idx] = v * inv_c;
      } else {
        v += yc_in[idx];
      }
      float av = fabsf(v) - lv[r];
      Zout[idx] = (av > 0.0f) ? copysignf(av, v) : 0.0f;
    }
  }
}

// ---------------------------------------------------------------------------
extern "C" void kernel_launch(void* const* d_in, const int* in_sizes, int n_in,
                              void* d_out, int out_size, void* d_ws, size_t ws_size,
                              hipStream_t stream) {
  const float* x     = (const float*)d_in[0];
  const float* Dict  = (const float*)d_in[1];
  const float* w1    = (const float*)d_in[2];
  const float* b1    = (const float*)d_in[3];
  const float* g1    = (const float*)d_in[4];
  const float* beta1 = (const float*)d_in[5];
  const float* w2    = (const float*)d_in[6];
  const float* b2    = (const float*)d_in[7];
  const float* g2    = (const float*)d_in[8];
  const float* beta2 = (const float*)d_in[9];
  const float* w3    = (const float*)d_in[10];
  const float* b3    = (const float*)d_in[11];
  float* out = (float*)d_out;

  char* ws = (char*)d_ws;
  size_t off = 0;
  auto take = [&](size_t bytes) {
    size_t o = off;
    off = (off + bytes + 255) & ~(size_t)255;
    return o;
  };
  const size_t zbytes = (size_t)NROWS * NATOMS * sizeof(float);
  float*  yc    = (float*)(ws + take(zbytes));
  float*  zA    = (float*)(ws + take(zbytes));
  float*  lthr  = (float*)(ws + take((size_t)NROWS * sizeof(float)));
  float*  G     = (float*)(ws + take((size_t)NATOMS * NATOMS * sizeof(float)));
  float*  cbuf  = (float*)(ws + take(256));
  __bf16* Dsw   = (__bf16*)(ws + take((size_t)LDIM * NATOMS * sizeof(__bf16)));
  __bf16* Ssw   = (__bf16*)(ws + take((size_t)NATOMS * NATOMS * sizeof(__bf16)));
  __bf16* w1sw  = (__bf16*)(ws + take((size_t)LDIM * 128 * sizeof(__bf16)));
  (void)ws_size; (void)in_sizes; (void)n_in; (void)out_size;

  const int MB = (NROWS + 63) / 64;   // 782

  gram_kernel<<<NATOMS, 256, 0, stream>>>(Dict, G);
  power_kernel<<<1, 512, 0, stream>>>(G, cbuf);

  pack_b_sw<<<(LDIM * NATOMS + 255) / 256, 256, 0, stream>>>(Dict, Dsw, LDIM, NATOMS, 0, cbuf);
  pack_b_sw<<<(NATOMS * NATOMS + 255) / 256, 256, 0, stream>>>(G, Ssw, NATOMS, NATOMS, 1, cbuf);
  pack_b_sw<<<(LDIM * 128 + 255) / 256, 256, 0, stream>>>(w1, w1sw, LDIM, 128, 0, cbuf);

  mlp_kernel<<<MB, 256, 0, stream>>>(x, w1sw, b1, g1, beta1, w2, b2, g2, beta2,
                                     w3, b3, cbuf, lthr, NROWS);

  dim3 grid(NATOMS / 256, MB);
  // y = x@Dict; store yc = y/c and z0 = soft(y, l) -> zA
  lista_gemm<<<grid, 256, 0, stream>>>(x, LDIM, LDIM, Dsw, lthr, cbuf,
                                       nullptr, yc, zA, NROWS, MODE_FIRST);
  // 5 iterations ping-pong; T=5 (odd) lands the final z in d_out.
  float* zin = zA;
  float* zout = out;
  for (int it = 0; it < TITERS; ++it) {
    lista_gemm<<<grid, 256, 0, stream>>>(zin, NATOMS, NATOMS, Ssw, lthr, cbuf,
                                         yc, nullptr, zout, NROWS, MODE_ITER);
    float* tmp = zin; zin = zout; zout = tmp;
  }
}
